// LSTM_10763188044500
// MI455X (gfx1250) — matrix operand order repaired and checked by reading, not compiled, not run on hardware
//
#include <hip/hip_runtime.h>
#include <hip/hip_bf16.h>

// ---------------- problem constants ----------------
#define B_   64
#define T_   2048
#define I_   512
#define H_   512
#define K_   1024            // I + H (concatenated GEMM K)
#define NG   2048            // 4*H gate columns
#define BH   (B_ * H_)

// ---------------- tiling ----------------
#define NBLK  64             // 32 hidden slices x 2 batch halves
#define MB    32             // batch rows per block
#define NH    16             // hidden units per block
#define KC    512            // K-chunk staged in LDS (x part / h part)
#define LDA   (KC + 16)      // LDS A row stride (bf16 elems), pad vs bank conflicts

typedef __bf16 bf16_t;
typedef __attribute__((ext_vector_type(16))) __bf16 v16bf;
typedef __attribute__((ext_vector_type(8)))  __bf16 v8bf;
typedef __attribute__((ext_vector_type(8)))  float  v8f;

__device__ __forceinline__ unsigned short bf_bits(float f) {
    union { float f; unsigned u; } v; v.f = f;
    unsigned r = v.u + 0x7FFFu + ((v.u >> 16) & 1u);   // round-to-nearest-even
    return (unsigned short)(r >> 16);
}
__device__ __forceinline__ bf16_t f2bf(float f) {
    unsigned short b = bf_bits(f);
    return __builtin_bit_cast(bf16_t, b);
}
__device__ __forceinline__ unsigned pack2(float a, float b) {
    return (unsigned)bf_bits(a) | ((unsigned)bf_bits(b) << 16);
}
__device__ __forceinline__ float sigmoidf_(float x) {
    return 1.0f / (1.0f + __expf(-x));
}

// ------------------------------------------------------------------
// Pack W = [W_i ; W_h] (K=1024) for gates [i|f|g|o] into bf16, N-major:
//   Wt[n][k], n = gate*512 + j, row stride K_.  bias[n] = b_i + b_h.
// ------------------------------------------------------------------
__global__ __launch_bounds__(256) void lstm_prep_weights(
    const float* __restrict__ wii, const float* __restrict__ whi,
    const float* __restrict__ bii, const float* __restrict__ bhi,
    const float* __restrict__ wif, const float* __restrict__ whf,
    const float* __restrict__ bif, const float* __restrict__ bhf,
    const float* __restrict__ wig, const float* __restrict__ whg,
    const float* __restrict__ big, const float* __restrict__ bhg,
    const float* __restrict__ wio, const float* __restrict__ who,
    const float* __restrict__ bio, const float* __restrict__ bho,
    bf16_t* __restrict__ Wt, float* __restrict__ biasC)
{
    int idx = blockIdx.x * blockDim.x + threadIdx.x;   // over NG*K_
    if (idx >= NG * K_) return;
    int n = idx >> 10;          // gate column (0..2047)
    int k = idx & (K_ - 1);
    int g = n >> 9;             // 0=i 1=f 2=g 3=o
    int j = n & (H_ - 1);

    float v;
    if (k < I_) {
        const float* w = (g == 0) ? wii : (g == 1) ? wif : (g == 2) ? wig : wio;
        v = w[k * H_ + j];
    } else {
        const float* w = (g == 0) ? whi : (g == 1) ? whf : (g == 2) ? whg : who;
        v = w[(k - I_) * H_ + j];
    }
    Wt[idx] = f2bf(v);

    if (k == 0) {
        const float* bi = (g == 0) ? bii : (g == 1) ? bif : (g == 2) ? big : bio;
        const float* bh = (g == 0) ? bhi : (g == 1) ? bhf : (g == 2) ? bhg : bho;
        biasC[n] = bi[j] + bh[j];
    }
}

// ------------------------------------------------------------------
// Persistent LSTM: 64 blocks loop over all T steps with a software
// device-wide barrier between steps.  Per step each block computes a
// (MB x 4*NH) slice of the gate pre-activations with bf16 WMMA
// (K = 1024 = [x_t | h_{t-1}]) and applies the cell update locally.
// Two independent WMMA accumulator chains hide matrix-op latency;
// bias is folded into the accumulator init (one value per lane).
// ------------------------------------------------------------------
__global__ __launch_bounds__(256) void lstm_persistent(
    const float*  __restrict__ x,       // (B,T,I)
    const bf16_t* __restrict__ Wt,      // (NG, K_) packed bf16
    const float*  __restrict__ biasC,   // (NG)
    float* __restrict__ hbuf,           // 2 * BH  (ping-pong)
    float* __restrict__ cbuf,           // BH      (in-place, owner-computes)
    float* __restrict__ out,            // h (BH) then c (BH)
    unsigned* __restrict__ bar_cnt,
    unsigned* __restrict__ bar_gen)
{
    __shared__ bf16_t ldsA[MB][LDA];
    __shared__ float  ldsP[MB][65];     // 4*NH = 64 preacts per row, +1 pad

    const int tid   = threadIdx.x;
    const int bx    = blockIdx.x;
    const int nb    = bx >> 1;          // hidden slice 0..31
    const int mhalf = bx & 1;           // batch half
    const int mBase = mhalf * MB;
    const int nBase = nb * NH;

    const int wave = tid >> 5;          // 0..7
    const int lane = tid & 31;
    const int mt   = wave >> 2;         // m-tile within block (0..1)
    const int gate = wave & 3;          // 0..3
    const int half = lane >> 4;         // 0: lanes 0-15, 1: lanes 16-31

    // B-fragment: 16 contiguous K per half-wave from row n = gate*H + j
    const int nGlob = gate * H_ + nBase + (lane & 15);
    const bf16_t* wrow = Wt + (size_t)nGlob * K_ + (half << 4);

    // Bias for this lane's output column (constant across t and rows):
    // C/D layout puts column n = lane&15 in every acc element of this lane.
    const float bias_l = biasC[nGlob];

    // A-fragment: lane holds row (lane&15)+16*mt; K 0..15
    // for lanes 0-15, 16..31 for lanes 16-31 (matches B fragment).
    const int arow = mt * 16 + (lane & 15);
    const int hi8  = half << 3;

    for (int t = 0; t < T_; ++t) {
        const float* hprev = hbuf + (t & 1) * BH;
        // acc0 carries the bias; acc1 is a second independent chain.
        v8f acc0 = {bias_l, bias_l, bias_l, bias_l, bias_l, bias_l, bias_l, bias_l};
        v8f acc1 = {0.f, 0.f, 0.f, 0.f, 0.f, 0.f, 0.f, 0.f};

        #pragma unroll
        for (int c = 0; c < 2; ++c) {
            __syncthreads();
            // Stage A chunk (bf16) into LDS: chunk 0 = x_t, chunk 1 = h_prev
            for (int q = tid; q < MB * (KC / 4); q += 256) {
                int r  = q >> 7;            // row (KC/4 = 128 quads per row)
                int kk = (q & 127) << 2;    // col within chunk
                int bg = mBase + r;
                const float* sp = (c == 0)
                    ? (x + ((size_t)bg * T_ + t) * I_ + kk)
                    : (hprev + (size_t)bg * H_ + kk);
                float4 v4 = *(const float4*)sp;
                unsigned* dst = (unsigned*)&ldsA[r][kk];
                dst[0] = pack2(v4.x, v4.y);
                dst[1] = pack2(v4.z, v4.w);
            }
            __syncthreads();

            const bf16_t* wp = wrow + c * KC;
            #pragma unroll 2
            for (int k0 = 0; k0 < KC; k0 += 64) {
                union { v16bf v; v8bf h[2]; } a0, a1;
                a0.h[0] = *(const v8bf*)&ldsA[arow][k0 + 2 * hi8];
                a0.h[1] = *(const v8bf*)&ldsA[arow][k0 + 8 + 2 * hi8];
                a1.h[0] = *(const v8bf*)&ldsA[arow][k0 + 32 + 2 * hi8];
                a1.h[1] = *(const v8bf*)&ldsA[arow][k0 + 40 + 2 * hi8];
                v16bf b0 = *(const v16bf*)(wp + k0);
                v16bf b1 = *(const v16bf*)(wp + k0 + 32);
                acc0 = __builtin_amdgcn_wmma_f32_16x16x32_bf16(
                    false, a0.v, false, b0, (short)0, acc0, false, false);
                acc1 = __builtin_amdgcn_wmma_f32_16x16x32_bf16(
                    false, a1.v, false, b1, (short)0, acc1, false, false);
            }
        }

        // Scatter (acc0 + acc1) to LDS preact tile (MB x 64)
        __syncthreads();
        {
            int mlo = mt * 16 + (half << 3);
            int nl  = gate * 16 + (lane & 15);
            #pragma unroll
            for (int r = 0; r < 8; ++r)
                ldsP[mlo + r][nl] = acc0[r] + acc1[r];
        }
        __syncthreads();

        // Cell update: MB*NH elems, owner-computes c in place
        for (int e = tid; e < MB * NH; e += 256) {
            int mb = e >> 4;
            int jl = e & 15;
            int bg = mBase + mb;
            int jg = nBase + jl;
            float ig = sigmoidf_(ldsP[mb][jl]);
            float fg = sigmoidf_(ldsP[mb][16 + jl]);
            float gg = tanhf(ldsP[mb][32 + jl]);
            float og = sigmoidf_(ldsP[mb][48 + jl]);
            size_t off = (size_t)bg * H_ + jg;
            float cnew = fg * cbuf[off] + ig * gg;
            float hnew = og * tanhf(cnew);
            if (t < T_ - 1) {
                cbuf[off] = cnew;
                hbuf[((t + 1) & 1) * BH + off] = hnew;
            } else {
                out[off]      = hnew;   // final h
                out[BH + off] = cnew;   // final c
            }
        }

        // ---- device-wide barrier between timesteps ----
        __threadfence();
        __syncthreads();
        if (tid == 0) {
            unsigned g = __hip_atomic_load(bar_gen, __ATOMIC_RELAXED,
                                           __HIP_MEMORY_SCOPE_AGENT);
            unsigned arrived = __hip_atomic_fetch_add(bar_cnt, 1u,
                                   __ATOMIC_ACQ_REL, __HIP_MEMORY_SCOPE_AGENT);
            if (arrived == NBLK - 1) {
                __hip_atomic_store(bar_cnt, 0u, __ATOMIC_RELAXED,
                                   __HIP_MEMORY_SCOPE_AGENT);
                __hip_atomic_fetch_add(bar_gen, 1u, __ATOMIC_RELEASE,
                                       __HIP_MEMORY_SCOPE_AGENT);
            } else {
                while (__hip_atomic_load(bar_gen, __ATOMIC_ACQUIRE,
                                         __HIP_MEMORY_SCOPE_AGENT) == g)
                    __builtin_amdgcn_s_sleep(1);
            }
        }
        __syncthreads();
    }
}

// ------------------------------------------------------------------
extern "C" void kernel_launch(void* const* d_in, const int* in_sizes, int n_in,
                              void* d_out, int out_size, void* d_ws, size_t ws_size,
                              hipStream_t stream) {
    const float* x   = (const float*)d_in[0];
    const float* wii = (const float*)d_in[1],  *whi = (const float*)d_in[2];
    const float* bii = (const float*)d_in[3],  *bhi = (const float*)d_in[4];
    const float* wif = (const float*)d_in[5],  *whf = (const float*)d_in[6];
    const float* bif = (const float*)d_in[7],  *bhf = (const float*)d_in[8];
    const float* wig = (const float*)d_in[9],  *whg = (const float*)d_in[10];
    const float* big = (const float*)d_in[11], *bhg = (const float*)d_in[12];
    const float* wio = (const float*)d_in[13], *who = (const float*)d_in[14];
    const float* bio = (const float*)d_in[15], *bho = (const float*)d_in[16];

    // workspace layout: [Wt bf16 4MB][bias 8KB][hbuf 256KB][cbuf 128KB][barrier]
    char* ws = (char*)d_ws;
    bf16_t*  Wt    = (bf16_t*)ws;
    float*   biasC = (float*)(ws + (size_t)NG * K_ * 2);
    float*   hbuf  = (float*)((char*)biasC + NG * 4);
    float*   cbuf  = hbuf + 2 * BH;
    unsigned* bar  = (unsigned*)(cbuf + BH);

    // zero h0, c0 and barrier state (required every call; ws is not re-poisoned)
    hipMemsetAsync(hbuf, 0, (size_t)(3 * BH) * 4 + 256, stream);

    lstm_prep_weights<<<(NG * K_ + 255) / 256, 256, 0, stream>>>(
        wii, whi, bii, bhi, wif, whf, bif, bhf,
        wig, whg, big, bhg, wio, who, bio, bho, Wt, biasC);

    lstm_persistent<<<NBLK, 256, 0, stream>>>(
        x, Wt, biasC, hbuf, cbuf, (float*)d_out, bar, bar + 1);
}
